// Attention_83141976916236
// MI455X (gfx1250) — matrix operand order repaired and checked by reading, not compiled, not run on hardware
//
#include <hip/hip_runtime.h>

#define DEV __device__ __forceinline__

typedef __attribute__((ext_vector_type(16))) __bf16 v16bf;
typedef __attribute__((ext_vector_type(8)))  float  v8f;
typedef __attribute__((ext_vector_type(4)))  unsigned int v4u;
typedef __attribute__((ext_vector_type(8)))  unsigned int v8u;

constexpr int cB = 2, cN = 2048, cC = 512, cH = 8, cD = 64;
constexpr int cF = 3 * cC;           // 1536
constexpr float cSCALE = 0.125f;     // D^-0.5 = 1/8

union F16x16 {
  v16bf v;
  v4u   q[2];
};

DEV unsigned short f2bf(float f) {
  unsigned int u = __float_as_uint(f);
  u += 0x7FFFu + ((u >> 16) & 1u);     // round-to-nearest-even
  return (unsigned short)(u >> 16);
}

// 16 bf16 gathered from two 16B chunks (p0 -> elems 0..7, p1 -> elems 8..15)
DEV v16bf load_frag_2x16B(const unsigned short* p0, const unsigned short* p1) {
  F16x16 f;
  f.q[0] = *reinterpret_cast<const v4u*>(p0);
  f.q[1] = *reinterpret_cast<const v4u*>(p1);
  return f.v;
}

DEV v8f wmma_bf16(v16bf a, v16bf b, v8f c) {
  return __builtin_amdgcn_wmma_f32_16x16x32_bf16(
      /*neg_a=*/false, a, /*neg_b=*/false, b,
      /*c_mod=*/(short)0, c, /*reuse_a=*/false, /*reuse_b=*/false);
}

DEV v8f zero8() {
  v8f z = {0.f, 0.f, 0.f, 0.f, 0.f, 0.f, 0.f, 0.f};
  return z;
}

// ---------------------------------------------------------------- TDM
// 2D tile load Global -> LDS via Tensor Data Mover (inline asm; portable
// across ROCm7.2 / amdgpu-toolchain builtin-arity differences).
// Loads tile_d1 rows x tile_d0 bf16 elements, row stride stride_d0 elements,
// into LDS with 4-DWORD padding every 16 DWORDs (row stride 80B for
// tile_d0 == 32 elements).
DEV void tdm_load_2d_bf16(unsigned lds_byte_off, const unsigned short* gptr,
                          unsigned tensor_d0, unsigned tensor_d1,
                          unsigned tile_d0, unsigned tile_d1,
                          unsigned stride_d0) {
  unsigned long long ga = (unsigned long long)(uintptr_t)gptr;
  v4u g0;
  g0[0] = 1u;                                          // count=1, user D#
  g0[1] = lds_byte_off;                                // lds_addr
  g0[2] = (unsigned)(ga & 0xFFFFFFFFu);                // global_addr lo
  g0[3] = (unsigned)((ga >> 32) & 0x01FFFFFFu) | 0x80000000u; // hi + type=2
  v8u g1;
  // data_size=1 (2B) | pad_enable | pad_interval=3 (16 DW) | pad_amount=3 (4 DW)
  g1[0] = (1u << 16) | (1u << 20) | (3u << 22) | (3u << 25);
  g1[1] = (tensor_d0 & 0xFFFFu) << 16;                 // tensor_dim0[15:0]
  g1[2] = ((tensor_d0 >> 16) & 0xFFFFu) | ((tensor_d1 & 0xFFFFu) << 16);
  g1[3] = ((tensor_d1 >> 16) & 0xFFFFu) | ((tile_d0 & 0xFFFFu) << 16);
  g1[4] = (tile_d1 & 0xFFFFu);                         // tile_dim1, tile_dim2=0
  g1[5] = stride_d0;                                   // tensor_dim0_stride lo32
  g1[6] = 0u;
  g1[7] = 0u;
  asm volatile("tensor_load_to_lds %0, %1" :: "s"(g0), "s"(g1) : "memory");
}

// ---------------------------------------------------------------- convert
__global__ void cvt_f32_bf16_x4(const float* __restrict__ in,
                                unsigned short* __restrict__ out, int n4) {
  int i = blockIdx.x * blockDim.x + threadIdx.x;
  if (i < n4) {
    const float4 f = reinterpret_cast<const float4*>(in)[i];
    unsigned int lo = (unsigned)f2bf(f.x) | ((unsigned)f2bf(f.y) << 16);
    unsigned int hi = (unsigned)f2bf(f.z) | ((unsigned)f2bf(f.w) << 16);
    reinterpret_cast<uint2*>(out)[i] = make_uint2(lo, hi);
  }
}

// ---------------------------------------------------------------- QKV GEMM
// X (4096 x 512) bf16 * Wqkv (512 x 1536) bf16. Block tile 256(M) x 64(N),
// each wave owns 32x64 (8 WMMAs / K-step). A tile staged by TDM, B tile
// transposed manually into LDS.
__global__ __launch_bounds__(256) void qkv_gemm_kernel(
    const unsigned short* __restrict__ Xb,
    const unsigned short* __restrict__ Wb,
    unsigned short* __restrict__ Qb,
    unsigned short* __restrict__ Kb,
    unsigned short* __restrict__ Vb) {
  __shared__ unsigned short ldsA[256 * 40];  // TDM-padded: row r at r*80 bytes
  __shared__ unsigned short ldsBt[64][40];   // [n][k]
  const int tid = threadIdx.x;
  const int wave = tid >> 5, lane = tid & 31;
  const int lhalf = lane >> 4, l15 = lane & 15;
  const int khalf8 = lhalf * 8;
  const int mblock = blockIdx.x * 256;
  const int nblock = blockIdx.y * 64;

  v8f acc[2][4];
#pragma unroll
  for (int f = 0; f < 2; ++f)
#pragma unroll
    for (int t = 0; t < 4; ++t) acc[f][t] = zero8();

  const unsigned ldsA_off = (unsigned)(uintptr_t)(&ldsA[0]);

  for (int k0 = 0; k0 < cC; k0 += 32) {
    __syncthreads();
    if (wave == 0) {
      tdm_load_2d_bf16(ldsA_off, Xb + (size_t)mblock * cC + k0,
                       /*tensor_d0=*/cC, /*tensor_d1=*/256,
                       /*tile_d0=*/32, /*tile_d1=*/256, /*stride_d0=*/cC);
    }
#pragma unroll
    for (int i = 0; i < 8; ++i) {            // stage B^T tile (32k x 64n)
      int idx = tid + i * 256;
      int k = idx >> 6, n = idx & 63;
      ldsBt[n][k] = Wb[(size_t)(k0 + k) * cF + nblock + n];
    }
    if (wave == 0) __builtin_amdgcn_s_wait_tensorcnt(0);
    __syncthreads();

    const unsigned short* ar0 = &ldsA[(wave * 32 + l15) * 40];
    const unsigned short* ar1 = &ldsA[(wave * 32 + 16 + l15) * 40];
    v16bf a0 = load_frag_2x16B(ar0 + khalf8, ar0 + 16 + khalf8);
    v16bf a1 = load_frag_2x16B(ar1 + khalf8, ar1 + 16 + khalf8);
    v16bf bb[4];
#pragma unroll
    for (int t = 0; t < 4; ++t) {
      const unsigned short* bp = &ldsBt[t * 16 + l15][lhalf * 8];
      bb[t] = load_frag_2x16B(bp, bp + 16);
    }
#pragma unroll
    for (int t = 0; t < 4; ++t) acc[0][t] = wmma_bf16(a0, bb[t], acc[0][t]);
#pragma unroll
    for (int t = 0; t < 4; ++t) acc[1][t] = wmma_bf16(a1, bb[t], acc[1][t]);
  }

#pragma unroll
  for (int f = 0; f < 2; ++f) {
    const int rbase = mblock + wave * 32 + f * 16 + lhalf * 8;
#pragma unroll
    for (int t = 0; t < 4; ++t) {
      const int col = nblock + t * 16 + l15;   // [0,1536)
      const int which = col >> 9;              // 0=Q 1=K 2=V
      const int rem = col & 511;
      const int h = rem >> 6, d = rem & 63;
      unsigned short* dst = which == 0 ? Qb : (which == 1 ? Kb : Vb);
      const float sc = which == 0 ? cSCALE : 1.0f;
#pragma unroll
      for (int e = 0; e < 8; ++e) {
        int r = rbase + e;
        int b = r >> 11;
        int n = r & (cN - 1);
        dst[(((size_t)b * cH + h) * cN + n) * cD + d] = f2bf(acc[f][t][e] * sc);
      }
    }
  }
}

// ---------------------------------------------------------------- attention
// Flash-style: each wave owns one 16-row Q tile, streams j in steps of 32.
__global__ __launch_bounds__(256) void attn_flash_kernel(
    const unsigned short* __restrict__ Qb,
    const unsigned short* __restrict__ Kb,
    const unsigned short* __restrict__ Vb,
    unsigned short* __restrict__ AO) {
  __shared__ unsigned short ldsVt[cD][40];        // V^T tile  [d][j]
  __shared__ unsigned short ldsP[8][16][40];      // per-wave P transpose

  const int bh = blockIdx.y;
  const int tid = threadIdx.x;
  const int wave = tid >> 5, lane = tid & 31;
  const int lhalf = lane >> 4, l15 = lane & 15;
  const int khalf8 = lhalf * 8;

  const unsigned short* Qp = Qb + (size_t)bh * cN * cD;
  const unsigned short* Kp = Kb + (size_t)bh * cN * cD;
  const unsigned short* Vp = Vb + (size_t)bh * cN * cD;

  const int row0 = blockIdx.x * 128 + wave * 16;
  const int mrow = row0 + l15;

  const unsigned short* qrow = Qp + (size_t)mrow * cD;
  v16bf aq0 = load_frag_2x16B(qrow + khalf8, qrow + 16 + khalf8);        // K=0..31
  v16bf aq1 = load_frag_2x16B(qrow + 32 + khalf8, qrow + 48 + khalf8);   // K=32..63

  v8f o[4];
  float mrun[8], lrun[8];
#pragma unroll
  for (int e = 0; e < 8; ++e) { mrun[e] = -3.0e38f; lrun[e] = 0.f; }
#pragma unroll
  for (int t = 0; t < 4; ++t) o[t] = zero8();

  for (int j0 = 0; j0 < cN; j0 += 32) {
    __syncthreads();
#pragma unroll
    for (int i = 0; i < 8; ++i) {            // stage V^T (32j x 64d)
      int idx = tid + i * 256;
      int j = idx >> 6, d = idx & 63;
      ldsVt[d][j] = Vp[(size_t)(j0 + j) * cD + d];
    }
    // prefetch next step's K/V rows into cache
    if (j0 + 32 < cN) {
      __builtin_prefetch(Vp + (size_t)(j0 + 32) * cD + tid * 8, 0, 1);
      __builtin_prefetch(Kp + (size_t)(j0 + 32) * cD + tid * 8, 0, 1);
    }
    __syncthreads();

    // S = Q K^T : two 16x16 tiles (cols j0..+15, j0+16..+31), K-dim = 64
    const unsigned short* krow0 = Kp + (size_t)(j0 + l15) * cD;
    const unsigned short* krow1 = krow0 + 16 * cD;
    const int dh = lhalf * 8;
    v16bf bk00 = load_frag_2x16B(krow0 + dh,      krow0 + dh + 16);
    v16bf bk01 = load_frag_2x16B(krow0 + 32 + dh, krow0 + 48 + dh);
    v16bf bk10 = load_frag_2x16B(krow1 + dh,      krow1 + dh + 16);
    v16bf bk11 = load_frag_2x16B(krow1 + 32 + dh, krow1 + 48 + dh);

    v8f s0 = zero8(), s1 = zero8();
    s0 = wmma_bf16(aq0, bk00, s0);
    s1 = wmma_bf16(aq0, bk10, s1);
    s0 = wmma_bf16(aq1, bk01, s0);
    s1 = wmma_bf16(aq1, bk11, s1);

    // online softmax (row = e + 8*lhalf; columns live across 16-lane halves)
    float p0[8], p1[8];
#pragma unroll
    for (int e = 0; e < 8; ++e) {
      float tmax = fmaxf(s0[e], s1[e]);
      tmax = fmaxf(tmax, __shfl_xor(tmax, 1, 16));
      tmax = fmaxf(tmax, __shfl_xor(tmax, 2, 16));
      tmax = fmaxf(tmax, __shfl_xor(tmax, 4, 16));
      tmax = fmaxf(tmax, __shfl_xor(tmax, 8, 16));
      float mnew = fmaxf(mrun[e], tmax);
      float corr = __expf(mrun[e] - mnew);
      float e0 = __expf(s0[e] - mnew);
      float e1 = __expf(s1[e] - mnew);
      float ps = e0 + e1;
      ps += __shfl_xor(ps, 1, 16);
      ps += __shfl_xor(ps, 2, 16);
      ps += __shfl_xor(ps, 4, 16);
      ps += __shfl_xor(ps, 8, 16);
      lrun[e] = lrun[e] * corr + ps;
      mrun[e] = mnew;
      o[0][e] *= corr; o[1][e] *= corr; o[2][e] *= corr; o[3][e] *= corr;
      p0[e] = e0; p1[e] = e1;
    }

    // transpose P (D-layout -> A-layout) through wave-private LDS
#pragma unroll
    for (int e = 0; e < 8; ++e) {
      ldsP[wave][lhalf * 8 + e][l15]      = f2bf(p0[e]);
      ldsP[wave][lhalf * 8 + e][l15 + 16] = f2bf(p1[e]);
    }
    asm volatile("s_wait_dscnt 0" ::: "memory");
    const unsigned short* prow = &ldsP[wave][l15][0];
    v16bf ap = load_frag_2x16B(prow + khalf8, prow + 16 + khalf8);

    // O += P * V  (4 d-tiles of 16): preload all B-frags, then WMMA burst
    v16bf bv[4];
#pragma unroll
    for (int t = 0; t < 4; ++t) {
      const unsigned short* vp = &ldsVt[t * 16 + l15][lhalf * 8];
      bv[t] = load_frag_2x16B(vp, vp + 16);
    }
#pragma unroll
    for (int t = 0; t < 4; ++t) o[t] = wmma_bf16(ap, bv[t], o[t]);
  }

  // normalize and store; AO is flat (B,H,N,D) == reference reshape to (B,N,C)
  unsigned short* aop = AO + (size_t)bh * cN * cD;
#pragma unroll
  for (int t = 0; t < 4; ++t) {
#pragma unroll
    for (int e = 0; e < 8; ++e) {
      int r = row0 + lhalf * 8 + e;
      int d = t * 16 + l15;
      aop[(size_t)r * cD + d] = f2bf(o[t][e] / lrun[e]);
    }
  }
}

// ---------------------------------------------------------------- proj GEMM
__global__ __launch_bounds__(256) void proj_gemm_kernel(
    const unsigned short* __restrict__ AO,
    const unsigned short* __restrict__ Wb,
    const float* __restrict__ bias,
    float* __restrict__ out) {
  __shared__ unsigned short ldsA[256 * 40];
  __shared__ unsigned short ldsBt[64][40];
  const int tid = threadIdx.x;
  const int wave = tid >> 5, lane = tid & 31;
  const int lhalf = lane >> 4, l15 = lane & 15;
  const int khalf8 = lhalf * 8;
  const int mblock = blockIdx.x * 256;
  const int nblock = blockIdx.y * 64;

  v8f acc[2][4];
#pragma unroll
  for (int f = 0; f < 2; ++f)
#pragma unroll
    for (int t = 0; t < 4; ++t) acc[f][t] = zero8();

  const unsigned ldsA_off = (unsigned)(uintptr_t)(&ldsA[0]);

  for (int k0 = 0; k0 < cC; k0 += 32) {
    __syncthreads();
    if (wave == 0) {
      tdm_load_2d_bf16(ldsA_off, AO + (size_t)mblock * cC + k0,
                       cC, 256, 32, 256, cC);
    }
#pragma unroll
    for (int i = 0; i < 8; ++i) {
      int idx = tid + i * 256;
      int k = idx >> 6, n = idx & 63;
      ldsBt[n][k] = Wb[(size_t)(k0 + k) * cC + nblock + n];
    }
    if (wave == 0) __builtin_amdgcn_s_wait_tensorcnt(0);
    __syncthreads();

    const unsigned short* ar0 = &ldsA[(wave * 32 + l15) * 40];
    const unsigned short* ar1 = &ldsA[(wave * 32 + 16 + l15) * 40];
    v16bf a0 = load_frag_2x16B(ar0 + khalf8, ar0 + 16 + khalf8);
    v16bf a1 = load_frag_2x16B(ar1 + khalf8, ar1 + 16 + khalf8);
    v16bf bb[4];
#pragma unroll
    for (int t = 0; t < 4; ++t) {
      const unsigned short* bp = &ldsBt[t * 16 + l15][lhalf * 8];
      bb[t] = load_frag_2x16B(bp, bp + 16);
    }
#pragma unroll
    for (int t = 0; t < 4; ++t) acc[0][t] = wmma_bf16(a0, bb[t], acc[0][t]);
#pragma unroll
    for (int t = 0; t < 4; ++t) acc[1][t] = wmma_bf16(a1, bb[t], acc[1][t]);
  }

#pragma unroll
  for (int f = 0; f < 2; ++f) {
    const int rbase = mblock + wave * 32 + f * 16 + lhalf * 8;
#pragma unroll
    for (int t = 0; t < 4; ++t) {
      const int col = nblock + t * 16 + l15;
      const float bv = bias[col];
#pragma unroll
      for (int e = 0; e < 8; ++e) {
        int r = rbase + e;
        out[(size_t)r * cC + col] = acc[f][t][e] + bv;
      }
    }
  }
}

// ---------------------------------------------------------------- launch
extern "C" void kernel_launch(void* const* d_in, const int* in_sizes, int n_in,
                              void* d_out, int out_size, void* d_ws, size_t ws_size,
                              hipStream_t stream) {
  (void)in_sizes; (void)n_in; (void)out_size; (void)ws_size;
  const float* x      = (const float*)d_in[0];
  const float* qkv_w  = (const float*)d_in[1];
  const float* proj_w = (const float*)d_in[2];
  const float* proj_b = (const float*)d_in[3];
  float* out = (float*)d_out;

  char* base = (char*)d_ws;
  size_t off = 0;
  auto carve = [&](size_t elems) -> unsigned short* {
    unsigned short* p = (unsigned short*)(base + off);
    off += (elems * sizeof(unsigned short) + 255) & ~(size_t)255;
    return p;
  };
  const size_t nX  = (size_t)cB * cN * cC;       // 2M
  const size_t nWq = (size_t)cC * cF;            // 768K
  const size_t nWp = (size_t)cC * cC;            // 256K
  const size_t nHD = (size_t)cB * cH * cN * cD;  // 2M
  unsigned short* xb  = carve(nX);
  unsigned short* wqb = carve(nWq);
  unsigned short* wpb = carve(nWp);
  unsigned short* Qb  = carve(nHD);
  unsigned short* Kb  = carve(nHD);
  unsigned short* Vb  = carve(nHD);
  unsigned short* AOb = carve(nHD);

  cvt_f32_bf16_x4<<<(int)((nX  / 4 + 255) / 256), 256, 0, stream>>>(x,      xb,  (int)(nX  / 4));
  cvt_f32_bf16_x4<<<(int)((nWq / 4 + 255) / 256), 256, 0, stream>>>(qkv_w,  wqb, (int)(nWq / 4));
  cvt_f32_bf16_x4<<<(int)((nWp / 4 + 255) / 256), 256, 0, stream>>>(proj_w, wpb, (int)(nWp / 4));

  qkv_gemm_kernel<<<dim3(cB * cN / 256, cF / 64), 256, 0, stream>>>(xb, wqb, Qb, Kb, Vb);
  attn_flash_kernel<<<dim3(cN / 128, cB * cH), 256, 0, stream>>>(Qb, Kb, Vb, AOb);
  proj_gemm_kernel<<<dim3(cB * cN / 256, cC / 64), 256, 0, stream>>>(AOb, wpb, proj_b, out);
}
